// BiSPAModule_71012989272639
// MI455X (gfx1250) — compile-verified, matrix operand order checked by reading, not run on hardware
//
#include <hip/hip_runtime.h>
#include <hip/hip_bf16.h>

// MI455X / gfx1250 implementation of the BiSPA module.
// All GEMM-shaped compute (QKV proj, attention score/context, out-proj, MLP,
// 3x3 convs via implicit im2col) runs through v_wmma_f32_16x16x32_f16.
// LDS staging uses CDNA5 async global->LDS copies (ASYNCcnt) when available.

typedef __attribute__((ext_vector_type(16))) _Float16 v16h;
typedef __attribute__((ext_vector_type(8)))  _Float16 v8h;
typedef __attribute__((ext_vector_type(8)))  float    v8f;
typedef __attribute__((ext_vector_type(4)))  int      v4i;

#define WMMA_F16F32(a,b,c) \
  __builtin_amdgcn_wmma_f32_16x16x32_f16(false,(a),false,(b),(short)0,(c),false,false)

// ---------------------------------------------------------------------------
// CDNA5 async global->LDS copy (GLOBAL_LOAD_ASYNC_TO_LDS_B128, ASYNCcnt).
// Builtin signature (probe-confirmed by diagnostics): param0 = <4 x i32>* in
// the global segment, param1 = LDS pointer, then imm offset + imm cpol.
// Per ISA 10.x the instruction offset applies to BOTH the global and LDS
// addresses, so one base + imm offsets {0,16,...} streams a contiguous chunk.
// Flat pointers: low 32 bits of an LDS flat address are the LDS offset.
// ---------------------------------------------------------------------------
#if __has_builtin(__builtin_amdgcn_global_load_async_to_lds_b128)
#define BISPA_HAVE_ASYNC_LDS 1
#define ASYNC_G2L_B128(gp, lp, off)                                           \
  __builtin_amdgcn_global_load_async_to_lds_b128(                             \
      (__attribute__((address_space(1))) v4i*)(unsigned long long)(gp),       \
      (__attribute__((address_space(3))) v4i*)(unsigned)(unsigned long long)(lp), \
      (off), 0)
__device__ __forceinline__ void wait_async_all() {
#if __has_builtin(__builtin_amdgcn_s_wait_asynccnt)
  __builtin_amdgcn_s_wait_asynccnt(0);
#else
  asm volatile("s_wait_asynccnt 0x0" ::: "memory");
#endif
}
#endif

// ---------------------------------------------------------------------------
// Fragment loader: 16x32 tile (rows m, cols k), row-major with leading dim ld.
// Per ISA 16-bit A-matrix layout: lanes 0-15 -> M=lane, K groups {0..7,16..23};
// lanes 16-31 -> K groups {8..15,24..31}. Two 16B chunks per lane.
// The same mapping serves B^T fragments (rows n, cols k).
// ---------------------------------------------------------------------------
__device__ __forceinline__ v16h load_frag(const _Float16* base, int ld) {
  int lane = threadIdx.x & 31;
  int m    = lane & 15;
  int kb   = (lane >> 4) << 3;            // 0 or 8
  const _Float16* p = base + m * ld + kb;
  v8h lo = *(const v8h*)(p);              // K = kb .. kb+7
  v8h hi = *(const v8h*)(p + 16);         // K = 16+kb .. 16+kb+7
  v16h f;
#pragma unroll
  for (int i = 0; i < 8; ++i) { f[i] = lo[i]; f[i + 8] = hi[i]; }
  return f;
}

// ---------------------------------------------------------------------------
// Generic WMMA GEMM: C[M,N] = A[M,K] * Bt[N,K]^T + bias, optional ReLU.
// 128 threads = 4 waves. Block tile 64x64, wave tile 32x32, K-step 32.
// ---------------------------------------------------------------------------
__global__ __launch_bounds__(128) void gemm_wmma(
    const _Float16* __restrict__ A, int lda,
    const _Float16* __restrict__ Bt, int ldb,
    const float* __restrict__ bias,
    _Float16* __restrict__ Cout, int ldc,
    int K, int relu)
{
  __shared__ _Float16 As[64 * 32];
  __shared__ _Float16 Bs[64 * 32];
  const int bm = blockIdx.x, bn = blockIdx.y;
  const int tid = threadIdx.x;
  const int wave = tid >> 5, lane = tid & 31;
  const int wm = (wave & 1) << 5;
  const int wn = (wave >> 1) << 5;
  v8f acc00, acc01, acc10, acc11;
#pragma unroll
  for (int r = 0; r < 8; ++r) { acc00[r] = 0.f; acc01[r] = 0.f; acc10[r] = 0.f; acc11[r] = 0.f; }

  const int lr = tid >> 1;                // 0..63 : tile row
  const int lc = (tid & 1) << 4;          // 0 or 16 : half-row of 32 halves
  for (int k0 = 0; k0 < K; k0 += 32) {
    const _Float16* ga = A  + (size_t)(bm * 64 + lr) * lda + k0 + lc;
    const _Float16* gb = Bt + (size_t)(bn * 64 + lr) * ldb + k0 + lc;
#ifdef BISPA_HAVE_ASYNC_LDS
    ASYNC_G2L_B128(ga, &As[lr * 32 + lc], 0);
    ASYNC_G2L_B128(ga, &As[lr * 32 + lc], 16);
    ASYNC_G2L_B128(gb, &Bs[lr * 32 + lc], 0);
    ASYNC_G2L_B128(gb, &Bs[lr * 32 + lc], 16);
    wait_async_all();
#else
    const v8h* sa = (const v8h*)ga;
    *(v8h*)&As[lr * 32 + lc]     = sa[0];
    *(v8h*)&As[lr * 32 + lc + 8] = sa[1];
    const v8h* sb = (const v8h*)gb;
    *(v8h*)&Bs[lr * 32 + lc]     = sb[0];
    *(v8h*)&Bs[lr * 32 + lc + 8] = sb[1];
#endif
    __syncthreads();
    if (k0 + 32 < K) {                    // pull next tiles toward L2/WGP$
      __builtin_prefetch(ga + 32, 0, 1);
      __builtin_prefetch(gb + 32, 0, 1);
    }
    v16h a0 = load_frag(&As[wm * 32], 32);
    v16h a1 = load_frag(&As[(wm + 16) * 32], 32);
    v16h b0 = load_frag(&Bs[wn * 32], 32);
    v16h b1 = load_frag(&Bs[(wn + 16) * 32], 32);
    acc00 = WMMA_F16F32(a0, b0, acc00);
    acc01 = WMMA_F16F32(a0, b1, acc01);
    acc10 = WMMA_F16F32(a1, b0, acc10);
    acc11 = WMMA_F16F32(a1, b1, acc11);
    __syncthreads();
  }

  // C layout: n = lane&15, m = vgpr + 8*(lane>=16)
  const int jn = lane & 15;
  const int mb = bm * 64 + wm + ((lane >> 4) << 3);
#pragma unroll
  for (int j = 0; j < 2; ++j) {
    int n = bn * 64 + wn + j * 16 + jn;
    float bv = bias ? bias[n] : 0.f;
    v8f c0 = j ? acc01 : acc00;
    v8f c1 = j ? acc11 : acc10;
#pragma unroll
    for (int r = 0; r < 8; ++r) {
      float u0 = c0[r] + bv, u1 = c1[r] + bv;
      if (relu) { u0 = fmaxf(u0, 0.f); u1 = fmaxf(u1, 0.f); }
      Cout[(size_t)(mb + r) * ldc + n]      = (_Float16)u0;
      Cout[(size_t)(mb + 16 + r) * ldc + n] = (_Float16)u1;
    }
  }
}

// ---------------------------------------------------------------------------
// Flash-style attention, one block per (sequence, head). 128 threads, 4 waves.
// qkv rows: token index, ld=768 halves (q|k|v each 256, head slice d=32).
// Horizontal: seq=(b,l), tokens stride 1, mask j+l>=256.
// Vertical:   seq=(b,w), tokens stride W,  mask j+w>=256.
// ---------------------------------------------------------------------------
__global__ __launch_bounds__(128) void attn_flash(
    const _Float16* __restrict__ qkv,
    _Float16* __restrict__ out,
    int T, int vertical)
{
  __shared__ _Float16 Qs[256 * 32];
  __shared__ _Float16 Ks[256 * 32];
  __shared__ _Float16 Vt[32 * 256];       // transposed V: [d][T]
  __shared__ _Float16 Ps[4][16 * 32];     // per-wave P-tile relayout scratch

  const int h = blockIdx.y;
  const int s = blockIdx.x;
  int base_tok, tok_stride, mofs;
  if (!vertical) { int b = s >> 8; int l = s & 255; base_tok = (b * 256 + l) * 128; tok_stride = 1;   mofs = l; }
  else           { int b = s >> 7; int w = s & 127; base_tok = b * 256 * 128 + w;   tok_stride = 128; mofs = w; }

  const int tid = threadIdx.x;
  for (int t = tid; t < T; t += 128) {
    size_t row = (size_t)(base_tok + t * tok_stride) * 768;
    const _Float16* q = qkv + row + h * 32;
    const _Float16* k = qkv + row + 256 + h * 32;
    const _Float16* v = qkv + row + 512 + h * 32;
#ifdef BISPA_HAVE_ASYNC_LDS
    ASYNC_G2L_B128(q, &Qs[t * 32], 0);
    ASYNC_G2L_B128(q, &Qs[t * 32], 16);
    ASYNC_G2L_B128(q, &Qs[t * 32], 32);
    ASYNC_G2L_B128(q, &Qs[t * 32], 48);
    ASYNC_G2L_B128(k, &Ks[t * 32], 0);
    ASYNC_G2L_B128(k, &Ks[t * 32], 16);
    ASYNC_G2L_B128(k, &Ks[t * 32], 32);
    ASYNC_G2L_B128(k, &Ks[t * 32], 48);
#else
    const v8h* qv = (const v8h*)q;
    const v8h* kv = (const v8h*)k;
    *(v8h*)&Qs[t * 32]      = qv[0]; *(v8h*)&Qs[t * 32 + 8]  = qv[1];
    *(v8h*)&Qs[t * 32 + 16] = qv[2]; *(v8h*)&Qs[t * 32 + 24] = qv[3];
    *(v8h*)&Ks[t * 32]      = kv[0]; *(v8h*)&Ks[t * 32 + 8]  = kv[1];
    *(v8h*)&Ks[t * 32 + 16] = kv[2]; *(v8h*)&Ks[t * 32 + 24] = kv[3];
#endif
    const v8h* vv = (const v8h*)v;
    v8h v0 = vv[0], v1 = vv[1], v2 = vv[2], v3 = vv[3];
#pragma unroll
    for (int c = 0; c < 8; ++c) {
      Vt[c * 256 + t]        = v0[c];
      Vt[(8 + c) * 256 + t]  = v1[c];
      Vt[(16 + c) * 256 + t] = v2[c];
      Vt[(24 + c) * 256 + t] = v3[c];
    }
  }
#ifdef BISPA_HAVE_ASYNC_LDS
  wait_async_all();
#endif
  __syncthreads();

  const int wave = tid >> 5, lane = tid & 31;
  const int jn = lane & 15;
  const float scale = 0.17677669529663687f;   // 1/sqrt(32)
  v8f zacc;
#pragma unroll
  for (int r = 0; r < 8; ++r) zacc[r] = 0.f;

  const int nmt = T >> 4;
  for (int mt = wave; mt < nmt; mt += 4) {    // wave-uniform: EXEC stays full
    v16h qf = load_frag(&Qs[mt * 16 * 32], 32);
    float rm[8], rs[8];
    v8f acc0, acc1;
#pragma unroll
    for (int r = 0; r < 8; ++r) { rm[r] = -1e30f; rs[r] = 0.f; acc0[r] = 0.f; acc1[r] = 0.f; }

    const int npairs = T >> 5;
    for (int pt = 0; pt < npairs; ++pt) {
      v16h kf0 = load_frag(&Ks[(pt * 32) * 32], 32);
      v16h kf1 = load_frag(&Ks[(pt * 32 + 16) * 32], 32);
      v8f sa = WMMA_F16F32(qf, kf0, zacc);
      v8f sb = WMMA_F16F32(qf, kf1, zacc);
#pragma unroll
      for (int r = 0; r < 8; ++r) {
        float va = sa[r] * scale;
        float vb = sb[r] * scale;
        int ja = pt * 32 + jn;
        if (ja + mofs >= 256)      va = -1e30f;
        if (ja + 16 + mofs >= 256) vb = -1e30f;
        float tmax = fmaxf(va, vb);
        tmax = fmaxf(tmax, __shfl_xor(tmax, 1, 32));
        tmax = fmaxf(tmax, __shfl_xor(tmax, 2, 32));
        tmax = fmaxf(tmax, __shfl_xor(tmax, 4, 32));
        tmax = fmaxf(tmax, __shfl_xor(tmax, 8, 32));
        float mnew = fmaxf(rm[r], tmax);
        float fac  = __expf(rm[r] - mnew);
        float pav  = __expf(va - mnew);
        float pbv  = __expf(vb - mnew);
        float psum = pav + pbv;
        psum += __shfl_xor(psum, 1, 32);
        psum += __shfl_xor(psum, 2, 32);
        psum += __shfl_xor(psum, 4, 32);
        psum += __shfl_xor(psum, 8, 32);
        rs[r] = rs[r] * fac + psum;
        rm[r] = mnew;
        acc0[r] *= fac;
        acc1[r] *= fac;
        int prow = r + ((lane >> 4) << 3);
        Ps[wave][prow * 32 + jn]      = (_Float16)pav;
        Ps[wave][prow * 32 + 16 + jn] = (_Float16)pbv;
      }
      // C-layout -> A-layout via per-wave LDS scratch (same-wave dep, no barrier)
      v16h pf  = load_frag(&Ps[wave][0], 32);
      v16h vt0 = load_frag(&Vt[0 * 256 + pt * 32], 256);
      v16h vt1 = load_frag(&Vt[16 * 256 + pt * 32], 256);
      acc0 = WMMA_F16F32(pf, vt0, acc0);
      acc1 = WMMA_F16F32(pf, vt1, acc1);
    }

    int mbase = mt * 16 + ((lane >> 4) << 3);
#pragma unroll
    for (int r = 0; r < 8; ++r) {
      int m = mbase + r;
      size_t trow = (size_t)(base_tok + m * tok_stride) * 256 + h * 32;
      out[trow + jn]      = (_Float16)(acc0[r] / rs[r]);
      out[trow + 16 + jn] = (_Float16)(acc1[r] / rs[r]);
    }
  }
}

// ---------------------------------------------------------------------------
// 3x3 conv (pad=1) as implicit-GEMM WMMA. Input/output NHWC f16 (f32 option).
// Weights pre-permuted to [tap][cout][cin] f16. Block: 64 pixels x 64 cout.
// ---------------------------------------------------------------------------
__global__ __launch_bounds__(128) void conv3x3_wmma(
    const _Float16* __restrict__ X, const _Float16* __restrict__ Wt,
    const float* __restrict__ bias, void* __restrict__ Yout, int out_f32)
{
  __shared__ _Float16 As[64 * 32];
  __shared__ _Float16 Bs[64 * 32];
  const int pb = blockIdx.x;              // B * L * (W/64) = 1024
  const int xb = pb & 1;
  const int y  = (pb >> 1) & 255;
  const int b  = pb >> 9;
  const int bn = blockIdx.y;              // cout block (0..3)
  const int x0 = xb * 64;
  const int tid = threadIdx.x;
  const int wave = tid >> 5, lane = tid & 31;
  const int wm = (wave & 1) << 5;
  const int wn = (wave >> 1) << 5;
  v8f acc00, acc01, acc10, acc11;
#pragma unroll
  for (int r = 0; r < 8; ++r) { acc00[r] = 0.f; acc01[r] = 0.f; acc10[r] = 0.f; acc11[r] = 0.f; }

  const int lr = tid >> 1;
  const int lc = (tid & 1) << 4;
  for (int tap = 0; tap < 9; ++tap) {
    int dy = tap / 3 - 1, dx = tap % 3 - 1;
    int ys = y + dy;
    bool yok = (unsigned)ys < 256u;
    for (int kc = 0; kc < 256; kc += 32) {
      int xs = x0 + lr + dx;
      if (yok && (unsigned)xs < 128u) {
        const _Float16* ga = X + (((size_t)(b * 256 + ys) * 128 + xs) * 256 + kc + lc);
#ifdef BISPA_HAVE_ASYNC_LDS
        ASYNC_G2L_B128(ga, &As[lr * 32 + lc], 0);
        ASYNC_G2L_B128(ga, &As[lr * 32 + lc], 16);
#else
        const v8h* sa = (const v8h*)ga;
        *(v8h*)&As[lr * 32 + lc]     = sa[0];
        *(v8h*)&As[lr * 32 + lc + 8] = sa[1];
#endif
      } else {
        v8h z;
#pragma unroll
        for (int i = 0; i < 8; ++i) z[i] = (_Float16)0.f;
        *(v8h*)&As[lr * 32 + lc]     = z;
        *(v8h*)&As[lr * 32 + lc + 8] = z;
      }
      const _Float16* gb = Wt + (((size_t)tap * 256 + bn * 64 + lr) * 256 + kc + lc);
#ifdef BISPA_HAVE_ASYNC_LDS
      ASYNC_G2L_B128(gb, &Bs[lr * 32 + lc], 0);
      ASYNC_G2L_B128(gb, &Bs[lr * 32 + lc], 16);
      wait_async_all();
#else
      const v8h* sb = (const v8h*)gb;
      *(v8h*)&Bs[lr * 32 + lc]     = sb[0];
      *(v8h*)&Bs[lr * 32 + lc + 8] = sb[1];
#endif
      __syncthreads();
      v16h a0 = load_frag(&As[wm * 32], 32);
      v16h a1 = load_frag(&As[(wm + 16) * 32], 32);
      v16h b0 = load_frag(&Bs[wn * 32], 32);
      v16h b1 = load_frag(&Bs[(wn + 16) * 32], 32);
      acc00 = WMMA_F16F32(a0, b0, acc00);
      acc01 = WMMA_F16F32(a0, b1, acc01);
      acc10 = WMMA_F16F32(a1, b0, acc10);
      acc11 = WMMA_F16F32(a1, b1, acc11);
      __syncthreads();
    }
  }

  const int jn = lane & 15;
  const int mrow = wm + ((lane >> 4) << 3);
#pragma unroll
  for (int j = 0; j < 2; ++j) {
    int n = bn * 64 + wn + j * 16 + jn;
    float bv = bias[n];
    v8f c0 = j ? acc01 : acc00;
    v8f c1 = j ? acc11 : acc10;
#pragma unroll
    for (int r = 0; r < 8; ++r) {
      float u0 = fmaxf(c0[r] + bv, 0.f);
      float u1 = fmaxf(c1[r] + bv, 0.f);
      int px0 = x0 + mrow + r;
      int px1 = px0 + 16;
      size_t o0 = ((size_t)(b * 256 + y) * 128 + px0) * 256 + n;
      size_t o1 = ((size_t)(b * 256 + y) * 128 + px1) * 256 + n;
      if (out_f32) { ((float*)Yout)[o0] = u0; ((float*)Yout)[o1] = u1; }
      else { ((_Float16*)Yout)[o0] = (_Float16)u0; ((_Float16*)Yout)[o1] = (_Float16)u1; }
    }
  }
}

// ---------------------------------------------------------------------------
// LayerNorm over C=256 per token; one wave per token (wave32 -> 8 vals/lane).
// ---------------------------------------------------------------------------
__global__ __launch_bounds__(256) void layernorm_kernel(
    const _Float16* __restrict__ x, const float* __restrict__ g,
    const float* __restrict__ bta, _Float16* __restrict__ yout, int ntok)
{
  int wave = threadIdx.x >> 5, lane = threadIdx.x & 31;
  int tok = blockIdx.x * 8 + wave;
  if (tok >= ntok) return;
  const _Float16* row = x + (size_t)tok * 256;
  float v[8];
  float sum = 0.f;
#pragma unroll
  for (int i = 0; i < 8; ++i) { v[i] = (float)row[lane + i * 32]; sum += v[i]; }
#pragma unroll
  for (int m = 16; m >= 1; m >>= 1) sum += __shfl_xor(sum, m, 32);
  float mean = sum * (1.f / 256.f);
  float var = 0.f;
#pragma unroll
  for (int i = 0; i < 8; ++i) { float d = v[i] - mean; var += d * d; }
#pragma unroll
  for (int m = 16; m >= 1; m >>= 1) var += __shfl_xor(var, m, 32);
  float rstd = rsqrtf(var * (1.f / 256.f) + 1e-5f);
#pragma unroll
  for (int i = 0; i < 8; ++i) {
    int c = lane + i * 32;
    yout[(size_t)tok * 256 + c] = (_Float16)((v[i] - mean) * rstd * g[c] + bta[c]);
  }
}

// ---------------------------------------------------------------------------
// TDM (Tensor Data Mover) path: DMA a 64x32 f16 tile into LDS via a D#
// descriptor (ISA ch.8), wait on TENSORcnt, then WMMA from the tile.
// Descriptor packing per D# group0/group1 tables (count=1, type=2,
// data_size=2B, tile 32x64, tensor_dim0_stride=256).
// ---------------------------------------------------------------------------
#if __has_builtin(__builtin_amdgcn_tensor_load_to_lds) && __has_builtin(__builtin_amdgcn_s_wait_tensorcnt)
typedef __attribute__((ext_vector_type(4))) unsigned int u32x4;
typedef __attribute__((ext_vector_type(8))) int i32x8;
typedef __attribute__((ext_vector_type(4))) int i32x4;

__global__ __launch_bounds__(32) void tdm_wmma_tile(
    const _Float16* __restrict__ A, float* __restrict__ C)
{
  __shared__ _Float16 tile[64 * 32];
  unsigned long long ga = (unsigned long long)A;
  unsigned lds = (unsigned)(unsigned long long)(&tile[0]);   // flat low 32b = LDS offset
  u32x4 g0;
  g0[0] = 1u;                                            // count=1 (valid, user mode)
  g0[1] = lds;                                           // lds_addr [63:32]
  g0[2] = (unsigned)(ga & 0xffffffffu);                  // global_addr [95:64]
  g0[3] = (unsigned)((ga >> 32) & 0x01ffffffu) | (2u << 30); // addr msbs | type=2
  i32x8 g1;
  g1[0] = 0x00010000;            // data_size=1 (2 bytes), no multicast/pad/iterate
  g1[1] = (int)(256u << 16);     // tensor_dim0 = 256 (bits 79:48, low part)
  g1[2] = (int)(4096u << 16);    // tensor_dim1 = 4096 (bits 111:80, low part)
  g1[3] = (int)(32u << 16);      // tile_dim0 = 32 (bits 127:112)
  g1[4] = 64;                    // tile_dim1 = 64 (bits 143:128)
  g1[5] = 256;                   // tensor_dim0_stride = 256 (bits 207:160)
  g1[6] = 0;
  g1[7] = 0;
  i32x4 g2; g2[0] = 0; g2[1] = 0; g2[2] = 0; g2[3] = 0;  // 2D tensor: groups 2/3 unused
  i32x4 g3 = g2;
#if __clang_major__ >= 23
  i32x8 g4;
#pragma unroll
  for (int i = 0; i < 8; ++i) g4[i] = 0;
  __builtin_amdgcn_tensor_load_to_lds(g0, g1, g2, g3, g4, 0);
#else
  __builtin_amdgcn_tensor_load_to_lds(g0, g1, g2, g3, 0);
#endif
  __builtin_amdgcn_s_wait_tensorcnt(0);
  __syncthreads();
  v16h a = load_frag(&tile[0], 32);
  v16h b = load_frag(&tile[32 * 32], 32);
  v8f c;
#pragma unroll
  for (int r = 0; r < 8; ++r) c[r] = 0.f;
  c = WMMA_F16F32(a, b, c);
  int lane = threadIdx.x & 31;
#pragma unroll
  for (int r = 0; r < 8; ++r)
    C[(size_t)((r + ((lane >> 4) << 3)) * 16 + (lane & 15))] = c[r];
}
#endif

// ---------------------------------------------------------------------------
// Elementwise helpers
// ---------------------------------------------------------------------------
__global__ void cvt_f32_f16(const float* __restrict__ s, _Float16* __restrict__ d, int n) {
  int i = blockIdx.x * blockDim.x + threadIdx.x;
  if (i < n) d[i] = (_Float16)s[i];
}

__global__ void conv_w_rearrange(const float* __restrict__ s, _Float16* __restrict__ d) {
  int i = blockIdx.x * blockDim.x + threadIdx.x;     // dst index over 9*256*256
  if (i >= 589824) return;
  int ci = i & 255, co = (i >> 8) & 255, tap = i >> 16;
  d[i] = (_Float16)s[(co * 256 + ci) * 9 + tap];     // OIHW -> [tap][co][ci]
}

__global__ void gather_kernel(const float* __restrict__ S, _Float16* __restrict__ xh) {
  size_t i4 = (size_t)blockIdx.x * blockDim.x + threadIdx.x;   // 4 channels/thread
  if (i4 >= 4194304u) return;
  size_t idx = i4 * 4;
  int c = (int)(idx & 255);
  int w = (int)((idx >> 8) & 127);
  int l = (int)((idx >> 15) & 255);
  int b = (int)(idx >> 23);
  int j = l + w; if (j > 255) j = 255;
  const float4 v = *(const float4*)(S + ((((size_t)(b * 256 + l)) * 256 + j) * 256 + c));
  xh[idx]     = (_Float16)v.x;
  xh[idx + 1] = (_Float16)v.y;
  xh[idx + 2] = (_Float16)v.z;
  xh[idx + 3] = (_Float16)v.w;
}

__global__ void zero_kernel(float* __restrict__ p, size_t n4) {
  size_t i = (size_t)blockIdx.x * blockDim.x + threadIdx.x;
  if (i < n4) *(float4*)(p + i * 4) = make_float4(0.f, 0.f, 0.f, 0.f);
}

__global__ void scatter_kernel(const float* __restrict__ Y, float* __restrict__ O) {
  size_t idx = (size_t)blockIdx.x * blockDim.x + threadIdx.x;  // B*L*W*C
  if (idx >= 16777216u) return;
  int c = (int)(idx & 255);
  int w = (int)((idx >> 8) & 127);
  int l = (int)((idx >> 15) & 255);
  int b = (int)(idx >> 23);
  int j = l + w; if (j > 255) j = 255;           // tri mask: j >= l always holds
  atomicAdd(O + ((((size_t)(b * 256 + l)) * 256 + j) * 256 + c), Y[idx]);
}

// ---------------------------------------------------------------------------
// Launcher
// ---------------------------------------------------------------------------
extern "C" void kernel_launch(void* const* d_in, const int* in_sizes, int n_in,
                              void* d_out, int out_size, void* d_ws, size_t ws_size,
                              hipStream_t stream)
{
  (void)in_sizes; (void)n_in; (void)out_size; (void)ws_size;
  const float* S       = (const float*)d_in[0];
  const float* h_in_w  = (const float*)d_in[1];
  const float* h_in_b  = (const float*)d_in[2];
  const float* h_out_w = (const float*)d_in[3];
  const float* h_out_b = (const float*)d_in[4];
  const float* v_in_w  = (const float*)d_in[5];
  const float* v_in_b  = (const float*)d_in[6];
  const float* v_out_w = (const float*)d_in[7];
  const float* v_out_b = (const float*)d_in[8];
  const float* mlp_w   = (const float*)d_in[9];
  const float* mlp_b   = (const float*)d_in[10];
  const float* ln_g    = (const float*)d_in[11];
  const float* ln_b    = (const float*)d_in[12];
  const float* conv1_w = (const float*)d_in[13];
  const float* conv1_b = (const float*)d_in[14];
  const float* conv2_w = (const float*)d_in[15];
  const float* conv2_b = (const float*)d_in[16];
  float* outS = (float*)d_out;

  char* ws = (char*)d_ws;
  _Float16* xh   = (_Float16*)(ws);                          //  33.5 MB  (65536 x 256 f16)
  _Float16* bufQ = (_Float16*)(ws + (size_t)33554432);       // 100.7 MB  (65536 x 768 f16)
  _Float16* bufA = (_Float16*)(ws + (size_t)134217728);      //  33.5 MB  (65536 x 256 f16)
  _Float16* Zb   = (_Float16*)(ws + (size_t)167772160);      //  67.1 MB  (65536 x 512 f16)
  _Float16* w_hin  = (_Float16*)(ws + (size_t)234881024);
  _Float16* w_vin  = w_hin  + 196608;
  _Float16* w_hout = w_vin  + 196608;
  _Float16* w_vout = w_hout + 65536;
  _Float16* w_mlp  = w_vout + 65536;
  _Float16* w_c1   = w_mlp  + 131072;
  _Float16* w_c2   = w_c1   + 589824;
  float* y2 = (float*)bufQ;                                  // conv2 f32 out reuses bufQ

  // Weight conversions to f16 (B^T layouts fall out of x @ W.T directly)
  cvt_f32_f16<<<768, 256, 0, stream>>>(h_in_w,  w_hin,  196608);
  cvt_f32_f16<<<768, 256, 0, stream>>>(v_in_w,  w_vin,  196608);
  cvt_f32_f16<<<256, 256, 0, stream>>>(h_out_w, w_hout, 65536);
  cvt_f32_f16<<<256, 256, 0, stream>>>(v_out_w, w_vout, 65536);
  cvt_f32_f16<<<512, 256, 0, stream>>>(mlp_w,   w_mlp,  131072);
  conv_w_rearrange<<<2304, 256, 0, stream>>>(conv1_w, w_c1);
  conv_w_rearrange<<<2304, 256, 0, stream>>>(conv2_w, w_c2);

  // Window gather S -> Sh (f16, NHWC tokens)
  gather_kernel<<<16384, 256, 0, stream>>>(S, xh);

  // Horizontal attention branch
  gemm_wmma<<<dim3(1024, 12), 128, 0, stream>>>(xh, 256, w_hin, 256, h_in_b, bufQ, 768, 256, 0);
  attn_flash<<<dim3(512, 8), 128, 0, stream>>>(bufQ, bufA, 128, 0);
  gemm_wmma<<<dim3(1024, 4), 128, 0, stream>>>(bufA, 256, w_hout, 256, h_out_b, Zb, 512, 256, 0);

  // Vertical attention branch (same token ordering, strided sequences)
  gemm_wmma<<<dim3(1024, 12), 128, 0, stream>>>(xh, 256, w_vin, 256, v_in_b, bufQ, 768, 256, 0);
  attn_flash<<<dim3(256, 8), 128, 0, stream>>>(bufQ, bufA, 256, 1);
  gemm_wmma<<<dim3(1024, 4), 128, 0, stream>>>(bufA, 256, w_vout, 256, v_out_b, Zb + 256, 512, 256, 0);

  // MLP (concat -> 256) + ReLU, then LayerNorm -> Sp (NHWC f16)
  gemm_wmma<<<dim3(1024, 4), 128, 0, stream>>>(Zb, 512, w_mlp, 512, mlp_b, bufA, 256, 512, 1);
  layernorm_kernel<<<8192, 256, 0, stream>>>(bufA, ln_g, ln_b, xh, 65536);

  // Two 3x3 convs (implicit-GEMM WMMA) with fused bias+ReLU
  conv3x3_wmma<<<dim3(1024, 4), 128, 0, stream>>>(xh,   w_c1, conv1_b, (void*)bufA, 0);
  conv3x3_wmma<<<dim3(1024, 4), 128, 0, stream>>>(bufA, w_c2, conv2_b, (void*)y2,   1);

  // Scatter-add back into S-shaped output (clip overlap -> atomics)
  zero_kernel<<<32768, 256, 0, stream>>>(outS, 8388608u);
  scatter_kernel<<<65536, 256, 0, stream>>>(y2, outS);
}
